// GCN_11579231830147
// MI455X (gfx1250) — matrix operand order repaired and checked by reading, not compiled, not run on hardware
//
#include <hip/hip_runtime.h>

// ---------------------------------------------------------------------------
// GCN layer for MI455X (gfx1250, wave32, WMMA):
//   h  = seq @ W^T + b          [B, N, 256]   (kernel 1, bf16 WMMA, K=256)
//   out= PReLU(adj @ h, alpha)  [B, N, 256]   (kernel 2, bf16 WMMA, K=4096)
// adj (268 MB fp32) is the HBM-bound stream and is read EXACTLY once
// (BN=256 -> one block per 128-row slab), non-temporal, double-buffered
// through LDS so global latency overlaps the WMMA pipe.  h is kept
// bf16-transposed in workspace (8 MB) so its tiles are contiguous and stay
// L2-resident (regular-temporal loads).
// ---------------------------------------------------------------------------

typedef __attribute__((ext_vector_type(16))) __bf16 v16bf;
typedef __attribute__((ext_vector_type(8)))  __bf16 v8bf;
typedef __attribute__((ext_vector_type(8)))  float  v8f;
typedef __attribute__((ext_vector_type(4)))  float  v4f;   // for NT b128 loads

#define IN_FT  256
#define OUT_FT 256
#define NNODE  4096
#define NBATCH 4

// kernel-2 tiling: 128(M) x 256(N) block tile, K panels of 32, double buffer
#define BM    128
#define BN2   256
#define KSTEP 32
#define LDP   40   // padded LDS row pitch in bf16 elements (80 B = 20 banks)

static __device__ __forceinline__ unsigned int bfbits(float x) {
  __bf16 h = (__bf16)x;                       // RNE cvt f32 -> bf16
  return (unsigned int)__builtin_bit_cast(unsigned short, h);
}

static __device__ __forceinline__ v8bf load8(const unsigned short* p) {
  return *reinterpret_cast<const v8bf*>(p);
}

static __device__ __forceinline__ v16bf cat16(v8bf lo, v8bf hi) {
  v16bf r;
#pragma unroll
  for (int i = 0; i < 8; ++i) { r[i] = lo[i]; r[8 + i] = hi[i]; }
  return r;
}

// pack 8 fp32 (two v4f) -> 8 bf16 in a uint4
static __device__ __forceinline__ uint4 pack8(v4f f0, v4f f1) {
  uint4 u;
  u.x = bfbits(f0[0]) | (bfbits(f0[1]) << 16);
  u.y = bfbits(f0[2]) | (bfbits(f0[3]) << 16);
  u.z = bfbits(f1[0]) | (bfbits(f1[1]) << 16);
  u.w = bfbits(f1[2]) | (bfbits(f1[3]) << 16);
  return u;
}

// ---------------------------------------------------------------------------
// Kernel 0: W (fp32 [256,256]) -> bf16 workspace copy (one-time, tiny)
// ---------------------------------------------------------------------------
__global__ __launch_bounds__(256) void gcn_wconv_kernel(
    const float* __restrict__ W, unsigned short* __restrict__ Wbf) {
  const int i = (blockIdx.x * 256 + threadIdx.x) * 4;   // 64 blocks x 256 thr x 4
  v4f f = *reinterpret_cast<const v4f*>(W + i);
  uint2 u;
  u.x = bfbits(f[0]) | (bfbits(f[1]) << 16);
  u.y = bfbits(f[2]) | (bfbits(f[3]) << 16);
  *reinterpret_cast<uint2*>(Wbf + i) = u;
}

// ---------------------------------------------------------------------------
// Kernel 1: h = seq @ W^T + b, stored bf16 TRANSPOSED: hT[b][o][m]
// One 16x16 output tile per wave, K=256 in 8 wmma_f32_16x16x32_bf16 steps.
// ---------------------------------------------------------------------------
__global__ __launch_bounds__(256) void gcn_linear_kernel(
    const float* __restrict__ seq, const unsigned short* __restrict__ Wbf,
    const float* __restrict__ bias, unsigned short* __restrict__ hT) {
  const int lane   = threadIdx.x & 31;
  const int wave   = threadIdx.x >> 5;
  const int gw     = blockIdx.x * 8 + wave;     // 16384 tiles total
  const int m0     = (gw >> 4) * 16;            // row tile in [0, B*N)
  const int o0     = (gw & 15) * 16;            // col tile in [0, 256)
  const int lane16 = lane & 15;
  const int laneHi = lane >> 4;

  v8f acc = {};
  const float* arow = seq + (size_t)(m0 + lane16) * IN_FT;

#pragma unroll
  for (int k0 = 0; k0 < IN_FT; k0 += 32) {
    // A fragment (16x32 bf16): lane holds K = base..base+7 and base+16..base+23,
    // base = k0 + laneHi*8  (ISA 16-bit A layout)
    const float* pa = arow + k0 + laneHi * 8;
    float fa[16];
    *reinterpret_cast<v4f*>(fa + 0)  = *reinterpret_cast<const v4f*>(pa + 0);
    *reinterpret_cast<v4f*>(fa + 4)  = *reinterpret_cast<const v4f*>(pa + 4);
    *reinterpret_cast<v4f*>(fa + 8)  = *reinterpret_cast<const v4f*>(pa + 16);
    *reinterpret_cast<v4f*>(fa + 12) = *reinterpret_cast<const v4f*>(pa + 20);
    v16bf a;
#pragma unroll
    for (int i = 0; i < 16; ++i) a[i] = (__bf16)fa[i];

    // B fragment (32x16 bf16): lane holds column o = o0+lane16,
    // K = base..base+7 and base+16..base+23, base = k0 + laneHi*8 (matches A).
    const unsigned short* pb = Wbf + (size_t)(o0 + lane16) * IN_FT + k0 + laneHi * 8;
    v16bf bm = cat16(load8(pb), load8(pb + 16));

    acc = __builtin_amdgcn_wmma_f32_16x16x32_bf16(
        false, a, false, bm, (short)0, acc, false, false);
  }

  // bias: C/D layout keeps N = o0+lane16 fixed per lane
  const float bb = bias[o0 + lane16];
#pragma unroll
  for (int r = 0; r < 8; ++r) acc[r] += bb;

  // store transposed: lane holds 8 consecutive rows m of one column o
  // -> one packed b128 store into hT[b][o][m]
  const int b     = m0 / NNODE;
  const int mloc0 = (m0 % NNODE) + laneHi * 8;
  uint4 u;
  u.x = bfbits(acc[0]) | (bfbits(acc[1]) << 16);
  u.y = bfbits(acc[2]) | (bfbits(acc[3]) << 16);
  u.z = bfbits(acc[4]) | (bfbits(acc[5]) << 16);
  u.w = bfbits(acc[6]) | (bfbits(acc[7]) << 16);
  unsigned short* pd = hT + ((size_t)(b * OUT_FT + o0 + lane16)) * NNODE + mloc0;
  *reinterpret_cast<uint4*>(pd) = u;
}

// ---------------------------------------------------------------------------
// Kernel 2: out = PReLU(adj @ h).
// 128x256 tile per 512-thread block (16 waves in a 4x4 grid; each wave owns
// 2x4 16x16 tiles).  adj fp32 streamed ONCE from HBM non-temporally,
// converted to bf16; hT bf16 tiles re-read from L2.  Double-buffered LDS:
// next panel's global loads issue before the current panel's WMMAs.
// ---------------------------------------------------------------------------
__global__ __launch_bounds__(512) void gcn_aggregate_kernel(
    const float* __restrict__ adj, const unsigned short* __restrict__ hT,
    const float* __restrict__ alpha_p, float* __restrict__ out) {
  __shared__ unsigned short sA[2][BM * LDP];    // adj panel  [m][k] bf16
  __shared__ unsigned short sB[2][BN2 * LDP];   // hT  panel  [n][k] bf16

  const int tid    = threadIdx.x;
  const int lane   = tid & 31;
  const int wave   = tid >> 5;          // 0..15
  const int lane16 = lane & 15;
  const int laneHi = lane >> 4;
  const int waveRow = wave & 3;         // rows waveRow*32 .. +31 (2 tiles)
  const int waveCol = wave >> 2;        // cols waveCol*64 .. +63 (4 tiles)

  const int b  = blockIdx.x >> 5;       // 128 blocks = 4 batches * 32 slabs
  const int m0 = (blockIdx.x & 31) * BM;

  const float*          adjB = adj + (size_t)b * NNODE * NNODE;
  const unsigned short* hTB  = hT  + (size_t)b * OUT_FT * NNODE;

  // adj staging: thread -> (row 0..127, k-quarter), 8 fp32 each
  const int aRow = tid >> 2;
  const int aOff = (tid & 3) * 8;
  // hT staging: thread -> (row 0..255, k-half), 16 bf16 each
  const int bRow = tid >> 1;
  const int bOff = (tid & 1) * 16;

  const float*          adjP = adjB + (size_t)(m0 + aRow) * NNODE + aOff;
  const unsigned short* hP   = hTB + (size_t)bRow * NNODE + bOff;

  unsigned short* sAp = &sA[0][aRow * LDP + aOff];  // buffer stride BM*LDP
  unsigned short* sBp = &sB[0][bRow * LDP + bOff];  // buffer stride BN2*LDP

  v8f acc[2][4];
#pragma unroll
  for (int i = 0; i < 2; ++i)
#pragma unroll
    for (int j = 0; j < 4; ++j) acc[i][j] = (v8f){};

  // ---- prologue: stage panel k0=0 into buffer 0 ----
  {
    v4f f0 = __builtin_nontemporal_load(reinterpret_cast<const v4f*>(adjP));
    v4f f1 = __builtin_nontemporal_load(reinterpret_cast<const v4f*>(adjP + 4));
    *reinterpret_cast<uint4*>(sAp) = pack8(f0, f1);
    uint4 h0 = *reinterpret_cast<const uint4*>(hP);
    uint4 h1 = *reinterpret_cast<const uint4*>(hP + 8);
    *reinterpret_cast<uint4*>(sBp)     = h0;
    *reinterpret_cast<uint4*>(sBp + 8) = h1;
  }
  __syncthreads();

  for (int k0 = 0; k0 < NNODE; k0 += KSTEP) {
    const int  cur     = (k0 >> 5) & 1;
    const int  nxt     = cur ^ 1;
    const bool hasNext = (k0 + KSTEP) < NNODE;

    // ---- issue next panel's global loads early (latency overlap) ----
    v4f  f0 = {}, f1 = {};
    uint4 h0 = {}, h1 = {};
    if (hasNext) {
      const float* ps = adjP + k0 + KSTEP;
      f0 = __builtin_nontemporal_load(reinterpret_cast<const v4f*>(ps));
      f1 = __builtin_nontemporal_load(reinterpret_cast<const v4f*>(ps + 4));
      const unsigned short* ph = hP + k0 + KSTEP;
      h0 = *reinterpret_cast<const uint4*>(ph);
      h1 = *reinterpret_cast<const uint4*>(ph + 8);
    }

    // ---- fragments from LDS (b128 loads, ISA A/B VGPR layouts) ----
    const unsigned short* sAc = sA[cur];
    const unsigned short* sBc = sB[cur];
    v16bf af[2];
#pragma unroll
    for (int rt = 0; rt < 2; ++rt) {
      const unsigned short* pa =
          sAc + (waveRow * 32 + rt * 16 + lane16) * LDP + laneHi * 8;
      af[rt] = cat16(load8(pa), load8(pa + 16));
    }
    v16bf bfrag[4];
#pragma unroll
    for (int ct = 0; ct < 4; ++ct) {
      const unsigned short* pb =
          sBc + (waveCol * 64 + ct * 16 + lane16) * LDP + laneHi * 8;
      bfrag[ct] = cat16(load8(pb), load8(pb + 16));
    }

#pragma unroll
    for (int rt = 0; rt < 2; ++rt)
#pragma unroll
      for (int ct = 0; ct < 4; ++ct)
        acc[rt][ct] = __builtin_amdgcn_wmma_f32_16x16x32_bf16(
            false, af[rt], false, bfrag[ct], (short)0, acc[rt][ct],
            false, false);

    // ---- convert + store next panel into the alternate buffer ----
    if (hasNext) {
      *reinterpret_cast<uint4*>(sAp + nxt * (BM * LDP)) = pack8(f0, f1);
      unsigned short* pd = sBp + nxt * (BN2 * LDP);
      *reinterpret_cast<uint4*>(pd)     = h0;
      *reinterpret_cast<uint4*>(pd + 8) = h1;
    }
    __syncthreads();   // one barrier per K panel (double buffer)
  }

  // ---- PReLU epilogue + non-temporal fp32 stores (write-once stream) ----
  const float alpha = *alpha_p;
  float* outB = out + (size_t)b * NNODE * OUT_FT;
#pragma unroll
  for (int rt = 0; rt < 2; ++rt) {
#pragma unroll
    for (int ct = 0; ct < 4; ++ct) {
      const int n = waveCol * 64 + ct * 16 + lane16;
#pragma unroll
      for (int r = 0; r < 8; ++r) {
        const int m = m0 + waveRow * 32 + rt * 16 + laneHi * 8 + r;
        const float v = acc[rt][ct][r];
        __builtin_nontemporal_store(v > 0.f ? v : alpha * v,
                                    outB + (size_t)m * OUT_FT + n);
      }
    }
  }
}

// ---------------------------------------------------------------------------
extern "C" void kernel_launch(void* const* d_in, const int* in_sizes, int n_in,
                              void* d_out, int out_size, void* d_ws, size_t ws_size,
                              hipStream_t stream) {
  const float* seq   = (const float*)d_in[0];
  const float* adj   = (const float*)d_in[1];
  const float* W     = (const float*)d_in[2];
  const float* bias  = (const float*)d_in[3];
  const float* alpha = (const float*)d_in[4];
  float* out = (float*)d_out;

  // workspace: [0, 128KB) W bf16 ; [128KB, +8MB) hT bf16 [B][256][4096]
  unsigned short* Wbf = (unsigned short*)d_ws;
  unsigned short* hT  = (unsigned short*)((char*)d_ws + (size_t)OUT_FT * IN_FT * 2);

  gcn_wconv_kernel<<<64, 256, 0, stream>>>(W, Wbf);
  gcn_linear_kernel<<<(NBATCH * NNODE / 16) * (OUT_FT / 16) / 8, 256, 0, stream>>>(
      seq, Wbf, bias, hT);
  gcn_aggregate_kernel<<<NBATCH * (NNODE / BM), 512, 0, stream>>>(
      adj, hT, alpha, out);
}
